// LinearFeatureAttention_40982577938859
// MI455X (gfx1250) — matrix operand order repaired and checked by reading, not compiled, not run on hardware
//
#include <hip/hip_runtime.h>
#include <hip/hip_bf16.h>

typedef __attribute__((ext_vector_type(16))) __bf16  v16bf;
typedef __attribute__((ext_vector_type(8)))  __bf16  v8bf;
typedef __attribute__((ext_vector_type(8)))  float   v8f;
typedef __attribute__((ext_vector_type(8)))  float   v8fv;

// C = act(A @ W + bias),  A: MxK fp32 row-major, W: KxN fp32 row-major, C: MxN fp32
// Block tile 128x64, BK=64, 256 threads = 8 waves; wave w -> 32x32 slab.
// Per barrier interval: 16 ds_load_b128 -> 8 v_wmma_f32_16x16x32_bf16 per wave.
__global__ __launch_bounds__(256)
void gemm_bias_act_kernel(const float* __restrict__ A,
                          const float* __restrict__ W,
                          const float* __restrict__ bias,
                          float* __restrict__ C,
                          int M, int N, int K, int act_elu) {
    __shared__ __align__(16) __bf16 lds_a[128 * 64];  // A tile, (row, k) row-major, 16 KB
    __shared__ __align__(16) __bf16 lds_bt[64 * 64];  // W tile, transposed (n, k),  8 KB

    const int t    = threadIdx.x;
    const int lane = t & 31;
    const int wave = t >> 5;

    const int blockRowBase = blockIdx.y * 128;  // M
    const int blockColBase = blockIdx.x * 64;   // N

    const int waveRow = (wave >> 1) * 32;       // 0,32,64,96
    const int waveCol = (wave & 1) * 32;        // 0 or 32

    // A-tile loader: 128 rows x 64 k, 32 floats per thread (2 threads/row)
    const int ar = t >> 1;
    const int ac = (t & 1) * 32;
    // W-tile loader (transpose-friendly): thread owns column n, 16 consecutive k.
    const int wn = t & 63;
    const int wk = (t >> 6) * 16;

    // WMMA fragment coords (wave32, 16-bit 16x32 layout per ISA tables)
    const int fm    = lane & 15;                // M (or N for B) within 16
    const int abase = (lane >> 4) * 8;          // A K sub-block 0 / 8
    const int kbase = (lane >> 4) * 8;          // B K sub-block 0 / 8

    v8f c00 = {}, c01 = {}, c10 = {}, c11 = {};

    const __bf16* a0p = lds_a  + (waveRow + fm)      * 64 + abase;
    const __bf16* a1p = lds_a  + (waveRow + 16 + fm) * 64 + abase;
    const __bf16* b0p = lds_bt + (waveCol + fm)      * 64 + kbase;
    const __bf16* b1p = lds_bt + (waveCol + 16 + fm) * 64 + kbase;

    for (int k0 = 0; k0 < K; k0 += 64) {
        // ---- stage A tile: 8x float4 -> 32 bf16, four b128 LDS stores ----
        {
            const float4* ap = (const float4*)(A + (size_t)(blockRowBase + ar) * K + k0 + ac);
            #pragma unroll
            for (int i = 0; i < 4; ++i) {
                float4 f0 = ap[2 * i];
                float4 f1 = ap[2 * i + 1];
                v8fv x = { f0.x, f0.y, f0.z, f0.w, f1.x, f1.y, f1.z, f1.w };
                *(v8bf*)(lds_a + ar * 64 + ac + 8 * i) = __builtin_convertvector(x, v8bf);
            }
            if (k0 + 64 < K) __builtin_prefetch(ap + (64 / 4), 0, 1);
        }
        // ---- stage W tile transposed: coalesced b32 column loads,
        //      packed bf16, two b128 LDS stores per thread ----
        {
            const float* wp = W + (size_t)(k0 + wk) * N + blockColBase + wn;
            float w[16];
            #pragma unroll
            for (int j = 0; j < 16; ++j) w[j] = wp[(size_t)j * N];
            v8fv y0 = { w[0], w[1], w[2],  w[3],  w[4],  w[5],  w[6],  w[7]  };
            v8fv y1 = { w[8], w[9], w[10], w[11], w[12], w[13], w[14], w[15] };
            *(v8bf*)(lds_bt + wn * 64 + wk)     = __builtin_convertvector(y0, v8bf);
            *(v8bf*)(lds_bt + wn * 64 + wk + 8) = __builtin_convertvector(y1, v8bf);
            if (k0 + 64 < K) __builtin_prefetch(wp + (size_t)64 * N, 0, 1);
        }
        __syncthreads();

        // ---- 2 K-substeps x (2x2 register tile) = 8 WMMA per barrier pair ----
        #pragma unroll
        for (int ks = 0; ks < 64; ks += 32) {
            v16bf fa0 = __builtin_shufflevector(*(const v8bf*)(a0p + ks), *(const v8bf*)(a0p + ks + 16),
                                                0, 1, 2, 3, 4, 5, 6, 7, 8, 9, 10, 11, 12, 13, 14, 15);
            v16bf fa1 = __builtin_shufflevector(*(const v8bf*)(a1p + ks), *(const v8bf*)(a1p + ks + 16),
                                                0, 1, 2, 3, 4, 5, 6, 7, 8, 9, 10, 11, 12, 13, 14, 15);
            v16bf fb0 = __builtin_shufflevector(*(const v8bf*)(b0p + ks), *(const v8bf*)(b0p + ks + 16),
                                                0, 1, 2, 3, 4, 5, 6, 7, 8, 9, 10, 11, 12, 13, 14, 15);
            v16bf fb1 = __builtin_shufflevector(*(const v8bf*)(b1p + ks), *(const v8bf*)(b1p + ks + 16),
                                                0, 1, 2, 3, 4, 5, 6, 7, 8, 9, 10, 11, 12, 13, 14, 15);

            c00 = __builtin_amdgcn_wmma_f32_16x16x32_bf16(false, fa0, false, fb0, (short)0, c00, false, false);
            c01 = __builtin_amdgcn_wmma_f32_16x16x32_bf16(false, fa0, false, fb1, (short)0, c01, false, false);
            c10 = __builtin_amdgcn_wmma_f32_16x16x32_bf16(false, fa1, false, fb0, (short)0, c10, false, false);
            c11 = __builtin_amdgcn_wmma_f32_16x16x32_bf16(false, fa1, false, fb1, (short)0, c11, false, false);
        }
        __syncthreads();
    }

    // ---- epilogue: bias + optional ELU ----
    const int n0 = blockColBase + waveCol + (lane & 15);
    const int m0 = blockRowBase + waveRow + ((lane < 16) ? 0 : 8);
    const float bias0 = bias[n0];
    const float bias1 = bias[n0 + 16];
    #pragma unroll
    for (int r = 0; r < 8; ++r) {
        float x00 = c00[r] + bias0;
        float x01 = c01[r] + bias1;
        float x10 = c10[r] + bias0;
        float x11 = c11[r] + bias1;
        if (act_elu) {
            x00 = (x00 > 0.0f) ? x00 : (__expf(x00) - 1.0f);
            x01 = (x01 > 0.0f) ? x01 : (__expf(x01) - 1.0f);
            x10 = (x10 > 0.0f) ? x10 : (__expf(x10) - 1.0f);
            x11 = (x11 > 0.0f) ? x11 : (__expf(x11) - 1.0f);
        }
        C[(size_t)(m0 + r) * N + n0]           = x00;
        C[(size_t)(m0 + r) * N + n0 + 16]      = x01;
        C[(size_t)(m0 + 16 + r) * N + n0]      = x10;
        C[(size_t)(m0 + 16 + r) * N + n0 + 16] = x11;
    }
}

// Per-row: qk = dot(Qf,Kf), sq = sum(Qf), sk = sum(Kf);
// out[row] = Vp[row] * qk / (sq*sk + eps).  F = 1024, D = 2048.
__global__ __launch_bounds__(256)
void finalize_kernel(const float* __restrict__ Qf,
                     const float* __restrict__ Kf,
                     const float* __restrict__ Vp,
                     float* __restrict__ out) {
    const int row = blockIdx.x;
    const int t   = threadIdx.x;

    const float4 q = ((const float4*)(Qf + (size_t)row * 1024))[t];
    const float4 k = ((const float4*)(Kf + (size_t)row * 1024))[t];

    float qk = q.x * k.x + q.y * k.y + q.z * k.z + q.w * k.w;
    float sq = q.x + q.y + q.z + q.w;
    float sk = k.x + k.y + k.z + k.w;

    __shared__ float r0[256], r1[256], r2[256];
    r0[t] = qk; r1[t] = sq; r2[t] = sk;
    __syncthreads();
    #pragma unroll
    for (int s = 128; s > 0; s >>= 1) {
        if (t < s) { r0[t] += r0[t + s]; r1[t] += r1[t + s]; r2[t] += r2[t + s]; }
        __syncthreads();
    }
    const float scale = r0[0] / (r1[0] * r2[0] + 1e-6f);

    const float4* v4 = (const float4*)(Vp + (size_t)row * 2048);
    float4*       o4 = (float4*)(out + (size_t)row * 2048);
    float4 a = v4[t];
    a.x *= scale; a.y *= scale; a.z *= scale; a.w *= scale;
    o4[t] = a;
    float4 b = v4[t + 256];
    b.x *= scale; b.y *= scale; b.z *= scale; b.w *= scale;
    o4[t + 256] = b;
}

extern "C" void kernel_launch(void* const* d_in, const int* in_sizes, int n_in,
                              void* d_out, int out_size, void* d_ws, size_t ws_size,
                              hipStream_t stream) {
    const int B = 8192, D = 2048, F = 1024;

    const float* Q  = (const float*)d_in[0];
    const float* K  = (const float*)d_in[1];
    const float* V  = (const float*)d_in[2];
    const float* Wq = (const float*)d_in[3];
    const float* bq = (const float*)d_in[4];
    const float* Wk = (const float*)d_in[5];
    const float* bk = (const float*)d_in[6];
    const float* Wv = (const float*)d_in[7];
    const float* bv = (const float*)d_in[8];

    // workspace layout: Qf (B*F) | Kf (B*F) | Vp (B*D)  -> 128 MiB total
    float* Qf = (float*)d_ws;
    float* Kf = Qf + (size_t)B * F;
    float* Vp = Kf + (size_t)B * F;

    dim3 blk(256);
    gemm_bias_act_kernel<<<dim3(F / 64, B / 128), blk, 0, stream>>>(Q, Wq, bq, Qf, B, F, D, 1);
    gemm_bias_act_kernel<<<dim3(F / 64, B / 128), blk, 0, stream>>>(K, Wk, bk, Kf, B, F, D, 1);
    gemm_bias_act_kernel<<<dim3(D / 64, B / 128), blk, 0, stream>>>(V, Wv, bv, Vp, B, D, D, 0);

    finalize_kernel<<<dim3(B), blk, 0, stream>>>(Qf, Kf, Vp, (float*)d_out);
}
